// CRsAEDense_47029891891289
// MI455X (gfx1250) — matrix operand order repaired and checked by reading, not compiled, not run on hardware
//
#include <hip/hip_runtime.h>
#include <math.h>

typedef __attribute__((ext_vector_type(16))) _Float16 v16h;
typedef __attribute__((ext_vector_type(8)))  _Float16 v8h;
typedef __attribute__((ext_vector_type(8)))  float    v8f;

#define BATCH   2048
#define D_IN    1024
#define D_ENC   4096
#define T_STEPS 30
#define INV_L      0.1f    // 1 / L, L = 10
#define LAM_OVER_L 0.01f   // lam / L = 0.1 / 10

// ---------------------------------------------------------------------------
// Pack a row-major f32 matrix into CDNA5 WMMA B-fragment order (f16).
// Logical B is [K][N]; tile (kt,nt) covers K in [kt*32,kt*32+32), N in [nt*16,+16).
// Packed element e: tile t = e>>9 (nt = t/numKt, kt = t%numKt),
//   lane = (e>>4)&31, slot s = e&15  ->  B[kt*32 + (lane>=16)*8 + s + (s>=8)*8][nt*16 + (lane&15)]
// This matches the A fragment K order: lanes 0-15 hold K=0..7,16..23,
// lanes 16-31 hold K=8..15,24..31, column n = lane%16.
// transpose==1: B[k][n] = src[n*K + k]  (src is H[1024][4096], B = H^T)
// transpose==0: B[k][n] = src[k*N + n]  (src is H, B = H)
// ---------------------------------------------------------------------------
__global__ __launch_bounds__(256) void fista_pack_b_kernel(
    const float* __restrict__ src, _Float16* __restrict__ dst,
    int K, int N, int transpose)
{
  const size_t e  = (size_t)blockIdx.x * blockDim.x + threadIdx.x;
  const int s     = (int)(e & 15);
  const int lane  = (int)((e >> 4) & 31);
  const size_t t  = e >> 9;
  const int numKt = K >> 5;
  const int nt    = (int)(t / numKt);
  const int kt    = (int)(t % numKt);
  const int k     = (kt << 4 << 1) + ((lane >= 16) ? 8 : 0) + s + ((s >= 8) ? 8 : 0);
  const int n     = (nt << 4) + (lane & 15);
  const float v   = transpose ? src[(size_t)n * K + k] : src[(size_t)k * N + n];
  dst[e] = (_Float16)v;
}

// A fragment (16x32 f16, M x K): lane holds row m = base + lane%16,
// halves 0..7  = K [k+b .. k+b+7],  halves 8..15 = K [k+b+16 .. k+b+23],
// with b = (lane>=16) ? 8 : 0.  Two aligned 16-byte loads.
__device__ __forceinline__ v16h fista_load_a_frag(
    const _Float16* __restrict__ A, int lda, int m, int k, int lane)
{
  const int b = (lane >= 16) ? 8 : 0;
  const _Float16* p = A + (size_t)(m + (lane & 15)) * lda + (k + b);
  union { v16h v; v8h h[2]; } u;
  u.h[0] = *(const v8h*)(p);
  u.h[1] = *(const v8h*)(p + 16);
  return u.v;
}

// ---------------------------------------------------------------------------
// One wave computes a 64x64 tile of C = A[MxK] * B[KxN] with f32 accumulation.
// MODE 0: residual   R16 = f16(Xin - C)                     (N = D_IN)
// MODE 1: FISTA      z = y + C/L; xn = shrink(z);
//                    yn = xn + momC*(xn - xold); writes Y16, Xnew16 (+f32 opt)
// MODE 2: final      Zout = C (f32)                          (N = D_IN)
// ---------------------------------------------------------------------------
template<int MODE>
__global__ __launch_bounds__(128) void fista_gemm_kernel(
    const _Float16* __restrict__ A, const _Float16* __restrict__ Bp,
    int M, int N, int K, int tilesM, int tilesN,
    const float* __restrict__ Xin, _Float16* __restrict__ R16,
    _Float16* __restrict__ Y16, const _Float16* __restrict__ Xold16,
    _Float16* __restrict__ Xnew16, float* __restrict__ Xout32,
    float* __restrict__ Zout, float momC)
{
  const int lane = threadIdx.x & 31;
  const int wave = blockIdx.x * (blockDim.x >> 5) + (threadIdx.x >> 5);
  const int tm = wave % tilesM;
  const int tn = wave / tilesM;
  if (tn >= tilesN) return;
  const int m0 = tm << 6;
  const int n0 = tn << 6;
  const int numKt = K >> 5;

  v8f acc[4][4];
#pragma unroll
  for (int i = 0; i < 4; ++i)
#pragma unroll
    for (int j = 0; j < 4; ++j)
#pragma unroll
      for (int v = 0; v < 8; ++v) acc[i][j][v] = 0.0f;

  for (int kt = 0; kt < numKt; ++kt) {
    const int k = kt << 5;
    v16h a[4];
#pragma unroll
    for (int i = 0; i < 4; ++i)
      a[i] = fista_load_a_frag(A, K, m0 + (i << 4), k, lane);
    v16h b[4];
#pragma unroll
    for (int j = 0; j < 4; ++j) {
      const _Float16* bp =
          Bp + ((size_t)((n0 >> 4) + j) * numKt + kt) * 512 + (lane << 4);
      b[j] = *(const v16h*)bp;   // 32B contiguous, pre-packed fragment
    }
#pragma unroll
    for (int i = 0; i < 4; ++i)
#pragma unroll
      for (int j = 0; j < 4; ++j)
        acc[i][j] = __builtin_amdgcn_wmma_f32_16x16x32_f16(
            false, a[i], false, b[j], (short)0, acc[i][j], false, false);
  }

  // C/D layout: lane -> column n = lane%16; v8f element v -> row m = 8*(lane>=16)+v
  const int col = lane & 15;
  const int rh  = (lane >> 4) << 3;
#pragma unroll
  for (int i = 0; i < 4; ++i) {
#pragma unroll
    for (int j = 0; j < 4; ++j) {
      const int n = n0 + (j << 4) + col;
#pragma unroll
      for (int v = 0; v < 8; ++v) {
        const int m = m0 + (i << 4) + rh + v;
        const size_t idx = (size_t)m * N + n;
        const float a = acc[i][j][v];
        if (MODE == 0) {
          R16[idx] = (_Float16)(Xin[idx] - a);
        } else if (MODE == 1) {
          const float y  = (float)Y16[idx];
          const float z  = fmaf(a, INV_L, y);
          const float az = fabsf(z) - LAM_OVER_L;
          const float xn = (az > 0.0f) ? copysignf(az, z) : 0.0f;
          const float xo = (float)Xold16[idx];
          const float yn = fmaf(momC, xn - xo, xn);
          Y16[idx]    = (_Float16)yn;
          Xnew16[idx] = (_Float16)xn;
          if (Xout32) Xout32[idx] = xn;
        } else {
          Zout[idx] = a;
        }
      }
    }
  }
}

extern "C" void kernel_launch(void* const* d_in, const int* in_sizes, int n_in,
                              void* d_out, int out_size, void* d_ws, size_t ws_size,
                              hipStream_t stream) {
  (void)in_sizes; (void)n_in; (void)out_size; (void)ws_size;
  const float* x32 = (const float*)d_in[0];   // [2048, 1024, 1] f32
  const float* H32 = (const float*)d_in[1];   // [1024, 4096]    f32
  float* out   = (float*)d_out;
  float* z_out = out;                               // 2048*1024 f32
  float* x_out = out + (size_t)BATCH * D_IN;        // 2048*4096 f32

  char* ws = (char*)d_ws;
  size_t off = 0;
  auto carve = [&](size_t bytes) -> void* {
    void* p = ws + off;
    off += (bytes + 255) & ~(size_t)255;
    return p;
  };
  const size_t HN  = (size_t)D_IN * D_ENC;          // 4 Mi elements
  const size_t YN  = (size_t)BATCH * D_ENC;         // 8 Mi elements
  const size_t RN  = (size_t)BATCH * D_IN;          // 2 Mi elements
  _Float16* HTp = (_Float16*)carve(sizeof(_Float16) * HN);  // B-frags of H^T [4096][1024]
  _Float16* Hp  = (_Float16*)carve(sizeof(_Float16) * HN);  // B-frags of H   [1024][4096]
  _Float16* y16 = (_Float16*)carve(sizeof(_Float16) * YN);
  _Float16* xa  = (_Float16*)carve(sizeof(_Float16) * YN);
  _Float16* xb  = (_Float16*)carve(sizeof(_Float16) * YN);
  _Float16* r16 = (_Float16*)carve(sizeof(_Float16) * RN);

  // Pack H into both B-fragment layouts (H is iteration-invariant).
  fista_pack_b_kernel<<<(unsigned)(HN / 256), 256, 0, stream>>>(H32, HTp, D_ENC, D_IN, 1);
  fista_pack_b_kernel<<<(unsigned)(HN / 256), 256, 0, stream>>>(H32, Hp,  D_IN, D_ENC, 0);

  // y0 = 0, x0 = 0
  hipMemsetAsync(y16, 0, sizeof(_Float16) * YN, stream);
  hipMemsetAsync(xa,  0, sizeof(_Float16) * YN, stream);

  const int tilesM  = BATCH / 64;   // 32
  const int tilesN1 = D_IN  / 64;   // 16
  const int tilesN2 = D_ENC / 64;   // 64
  const unsigned grid1 = (unsigned)(tilesM * tilesN1 / 4);  // 128 blocks
  const unsigned grid2 = (unsigned)(tilesM * tilesN2 / 4);  // 512 blocks

  float t = 1.0f;
  _Float16* xold = xa;
  _Float16* xnew = xb;
  for (int it = 0; it < T_STEPS; ++it) {
    // R = X - Y * H^T
    fista_gemm_kernel<0><<<grid1, 128, 0, stream>>>(
        y16, HTp, BATCH, D_IN, D_ENC, tilesM, tilesN1,
        x32, r16, nullptr, nullptr, nullptr, nullptr, nullptr, 0.0f);
    // FISTA momentum coefficients (deterministic scalar recurrence, host side)
    const float tnx = 0.5f * (1.0f + sqrtf(1.0f + 4.0f * t * t));
    const float c   = (t - 1.0f) / tnx;
    t = tnx;
    // Xnew = shrink(Y + (R*H)/L); Ynew = Xnew + c*(Xnew - Xold)
    float* xo32 = (it == T_STEPS - 1) ? x_out : nullptr;
    fista_gemm_kernel<1><<<grid2, 128, 0, stream>>>(
        r16, Hp, BATCH, D_ENC, D_IN, tilesM, tilesN2,
        nullptr, nullptr, y16, xold, xnew, xo32, nullptr, c);
    _Float16* tmp = xold; xold = xnew; xnew = tmp;
  }

  // z = Xfinal * H^T  (xold now points at the last-written X buffer)
  fista_gemm_kernel<2><<<grid1, 128, 0, stream>>>(
      xold, HTp, BATCH, D_IN, D_ENC, tilesM, tilesN1,
      nullptr, nullptr, nullptr, nullptr, nullptr, nullptr, z_out, 0.0f);
}
